// Generator_31396210933780
// MI455X (gfx1250) — compile-verified, matrix operand order checked by reading, not run on hardware
//
#include <hip/hip_runtime.h>
#include <hip/hip_bf16.h>

// Problem constants (match reference)
#define BB 32
#define TT 64
#define HH 1024
#define VV 32000
#define LL 10

typedef __attribute__((ext_vector_type(16))) __bf16 bfx16;
typedef __attribute__((ext_vector_type(8)))  __bf16 bfx8;
typedef __attribute__((ext_vector_type(8)))  float  fx8;

// ---------------------------------------------------------------------------
// Fragment loader: 16-bit A/B fragment for v_wmma_f32_16x16x32_bf16.
// Lane l: idx = l%16 selects the row (M for A, N for B), half = l/16 selects
// which K sub-blocks this lane holds. Per the CDNA5 ISA 16-bit A layout,
// a lane holds two contiguous 8-element (16-byte) chunks:
//   K in [k0 + 8*half, +8)      -> frag elements 0..7
//   K in [k0 + 16 + 8*half, +8) -> frag elements 8..15
// ---------------------------------------------------------------------------
__device__ __forceinline__ bfx16 load_frag_bf16(const __bf16* base, int row,
                                                int ldk, int k0, int half) {
    const __bf16* p = base + (size_t)row * ldk + k0 + 8 * half;
    bfx8 lo = *reinterpret_cast<const bfx8*>(p);
    bfx8 hi = *reinterpret_cast<const bfx8*>(p + 16);
    bfx16 r;
#pragma unroll
    for (int i = 0; i < 8; ++i) { r[i] = lo[i]; r[8 + i] = hi[i]; }
    return r;
}

__device__ __forceinline__ fx8 wmma_bf16(bfx16 a, bfx16 b, fx8 c) {
    return __builtin_amdgcn_wmma_f32_16x16x32_bf16(false, a, false, b,
                                                   (short)0, c, false, false);
}

// ---------------------------------------------------------------------------
// f32 -> bf16 conversion (one-shot per call; weights then live in L2 as bf16)
// ---------------------------------------------------------------------------
__global__ __launch_bounds__(256) void k_convert_bf16(const float* __restrict__ src,
                                                      __bf16* __restrict__ dst, int n) {
    int i = blockIdx.x * 256 + threadIdx.x;
    if (i < n) dst[i] = (__bf16)src[i];
}

// h0 = tanh(latent @ Wz_w^T + Wz_b); K = LL = 10 (tiny, plain VALU)
__global__ __launch_bounds__(256) void k_init_h0(const float* __restrict__ latent,
                                                 const float* __restrict__ Wz_w,
                                                 const float* __restrict__ Wz_b,
                                                 float* __restrict__ h,
                                                 __bf16* __restrict__ hbf) {
    int i = blockIdx.x * 256 + threadIdx.x;  // 0 .. B*H
    int b = i / HH, j = i % HH;
    float acc = Wz_b[j];
#pragma unroll
    for (int l = 0; l < LL; ++l) acc += latent[b * LL + l] * Wz_w[j * LL + l];
    float v = tanhf(acc);
    h[i] = v;
    hbf[i] = (__bf16)v;
}

// x0 = emb[input[:,0]] (gathered straight from f32 emb, converted to bf16)
__global__ __launch_bounds__(256) void k_init_x0(const int* __restrict__ input,
                                                 const float* __restrict__ emb,
                                                 __bf16* __restrict__ xbf) {
    int i = blockIdx.x * 256 + threadIdx.x;
    int b = i / HH, j = i % HH;
    xbf[i] = (__bf16)emb[(size_t)input[b] * HH + j];
}

// ---------------------------------------------------------------------------
// GRU gate GEMMs:  gi = x @ w_ih^T + b_ih,  gh = h @ w_hh^T + b_hh
// Each wave: 1 M-tile x 4 N-tiles (A-fragment reuse, 4 independent wmma
// chains). grid = (12, 2, 2): 192 N-tiles x 2 M-tiles x {gi, gh}.
// ---------------------------------------------------------------------------
__global__ __launch_bounds__(128) void k_gru_gates(const __bf16* __restrict__ xbf,
                                                   const __bf16* __restrict__ hbf,
                                                   const __bf16* __restrict__ wih,
                                                   const __bf16* __restrict__ whh,
                                                   const float* __restrict__ b_ih,
                                                   const float* __restrict__ b_hh,
                                                   float* __restrict__ gi,
                                                   float* __restrict__ gh) {
    const int wave = threadIdx.x >> 5;
    const int lane = threadIdx.x & 31;
    const int idx = lane & 15, half = lane >> 4;
    const int n0 = (blockIdx.x * 16 + wave * 4) * 16;  // 4 consecutive N tiles
    const int m0 = blockIdx.y * 16;
    const __bf16* A = (blockIdx.z == 0) ? xbf : hbf;
    const __bf16* W = (blockIdx.z == 0) ? wih : whh;
    const float* bias = (blockIdx.z == 0) ? b_ih : b_hh;
    float* out = (blockIdx.z == 0) ? gi : gh;

    fx8 acc0 = {}, acc1 = {}, acc2 = {}, acc3 = {};
#pragma unroll 2
    for (int k0 = 0; k0 < HH; k0 += 32) {
        bfx16 a  = load_frag_bf16(A, m0 + idx, HH, k0, half);
        bfx16 b0 = load_frag_bf16(W, n0 + idx,      HH, k0, half);
        bfx16 b1 = load_frag_bf16(W, n0 + 16 + idx, HH, k0, half);
        bfx16 b2 = load_frag_bf16(W, n0 + 32 + idx, HH, k0, half);
        bfx16 b3 = load_frag_bf16(W, n0 + 48 + idx, HH, k0, half);
        acc0 = wmma_bf16(a, b0, acc0);
        acc1 = wmma_bf16(a, b1, acc1);
        acc2 = wmma_bf16(a, b2, acc2);
        acc3 = wmma_bf16(a, b3, acc3);
    }
#pragma unroll
    for (int v = 0; v < 8; ++v) {
        int m = m0 + v + 8 * half;
        float* row = out + m * (3 * HH);
        int n = n0 + idx;
        row[n]      = acc0[v] + bias[n];
        row[n + 16] = acc1[v] + bias[n + 16];
        row[n + 32] = acc2[v] + bias[n + 32];
        row[n + 48] = acc3[v] + bias[n + 48];
    }
}

// Pointwise GRU update: h_new = (1-z)*n + z*h
__global__ __launch_bounds__(256) void k_gru_pointwise(const float* __restrict__ gi,
                                                       const float* __restrict__ gh,
                                                       float* __restrict__ h,
                                                       __bf16* __restrict__ hbf) {
    int i = blockIdx.x * 256 + threadIdx.x;  // 0 .. B*H
    int b = i / HH, j = i % HH;
    const float* gib = gi + b * 3 * HH;
    const float* ghb = gh + b * 3 * HH;
    float ir = gib[j], iz = gib[HH + j], in = gib[2 * HH + j];
    float hr = ghb[j], hz = ghb[HH + j], hn = ghb[2 * HH + j];
    float r = 1.0f / (1.0f + __expf(-(ir + hr)));
    float z = 1.0f / (1.0f + __expf(-(iz + hz)));
    float nn = tanhf(in + r * hn);
    float hv = (1.0f - z) * nn + z * h[i];
    h[i] = hv;
    hbf[i] = (__bf16)hv;
}

// ---------------------------------------------------------------------------
// Logits GEMM: logits = h @ out_w^T + out_b, written directly into d_out at
// rows (b*T + t). Each wave: 1 M-tile x 4 N-tiles with A reuse -> per k-step
// 10 b128 loads feed 4 independent v_wmma. grid = (125, 2): 2000 N-tiles
// (16 per block, 4 waves) x 2 M-tiles. ~65.5MB bf16 out_w streamed from L2
// per step (once per M-tile, so twice).
// ---------------------------------------------------------------------------
__global__ __launch_bounds__(128) void k_logits(const __bf16* __restrict__ hbf,
                                                const __bf16* __restrict__ W,
                                                const float* __restrict__ bias,
                                                float* __restrict__ out, int t) {
    const int wave = threadIdx.x >> 5;
    const int lane = threadIdx.x & 31;
    const int idx = lane & 15, half = lane >> 4;
    const int n0 = (blockIdx.x * 16 + wave * 4) * 16;  // 4 consecutive N tiles
    const int m0 = blockIdx.y * 16;

    fx8 acc0 = {}, acc1 = {}, acc2 = {}, acc3 = {};
#pragma unroll 2
    for (int k0 = 0; k0 < HH; k0 += 32) {
        bfx16 a  = load_frag_bf16(hbf, m0 + idx, HH, k0, half);
        bfx16 b0 = load_frag_bf16(W, n0 + idx,      HH, k0, half);
        bfx16 b1 = load_frag_bf16(W, n0 + 16 + idx, HH, k0, half);
        bfx16 b2 = load_frag_bf16(W, n0 + 32 + idx, HH, k0, half);
        bfx16 b3 = load_frag_bf16(W, n0 + 48 + idx, HH, k0, half);
        acc0 = wmma_bf16(a, b0, acc0);
        acc1 = wmma_bf16(a, b1, acc1);
        acc2 = wmma_bf16(a, b2, acc2);
        acc3 = wmma_bf16(a, b3, acc3);
    }
#pragma unroll
    for (int v = 0; v < 8; ++v) {
        int m = m0 + v + 8 * half;  // batch index
        float* row = out + ((size_t)(m * TT + t)) * VV;
        int n = n0 + idx;
        row[n]      = acc0[v] + bias[n];
        row[n + 16] = acc1[v] + bias[n + 16];
        row[n + 32] = acc2[v] + bias[n + 32];
        row[n + 48] = acc3[v] + bias[n + 48];
    }
}

// ---------------------------------------------------------------------------
// In-place log-softmax over V=32000 + argmax (feeds the next token).
// One workgroup per batch row; LDS tree reductions; ties -> lowest index.
// ---------------------------------------------------------------------------
__global__ __launch_bounds__(256) void k_logsoftmax_argmax(float* __restrict__ out,
                                                           int t, int* __restrict__ nxt) {
    const int b = blockIdx.x;
    float* row = out + ((size_t)(b * TT + t)) * VV;
    __shared__ float sred[256];
    __shared__ int sarg[256];
    const int tid = threadIdx.x;

    // pass 1: max + argmax
    float lmax = -3.402823466e38f;
    int larg = 0;
    for (int v = tid; v < VV; v += 256) {
        float x = row[v];
        if (x > lmax) { lmax = x; larg = v; }
    }
    sred[tid] = lmax; sarg[tid] = larg;
    __syncthreads();
    for (int s = 128; s > 0; s >>= 1) {
        if (tid < s) {
            float o = sred[tid + s]; int oa = sarg[tid + s];
            if (o > sred[tid] || (o == sred[tid] && oa < sarg[tid])) {
                sred[tid] = o; sarg[tid] = oa;
            }
        }
        __syncthreads();
    }
    float gmax = sred[0];
    int garg = sarg[0];
    __syncthreads();

    // pass 2: sum of exp
    float lsum = 0.0f;
    for (int v = tid; v < VV; v += 256) lsum += __expf(row[v] - gmax);
    sred[tid] = lsum;
    __syncthreads();
    for (int s = 128; s > 0; s >>= 1) {
        if (tid < s) sred[tid] += sred[tid + s];
        __syncthreads();
    }
    float c = gmax + __logf(sred[0]);

    // pass 3: normalize in place
    for (int v = tid; v < VV; v += 256) row[v] -= c;
    if (tid == 0) nxt[b] = garg;
}

// x = emb_bf16[argmax]
__global__ __launch_bounds__(256) void k_gather_x(const __bf16* __restrict__ embbf,
                                                  const int* __restrict__ nxt,
                                                  __bf16* __restrict__ xbf) {
    int i = blockIdx.x * 256 + threadIdx.x;
    int b = i / HH, j = i % HH;
    xbf[i] = embbf[(size_t)nxt[b] * HH + j];
}

// ---------------------------------------------------------------------------
extern "C" void kernel_launch(void* const* d_in, const int* in_sizes, int n_in,
                              void* d_out, int out_size, void* d_ws, size_t ws_size,
                              hipStream_t stream) {
    (void)in_sizes; (void)n_in; (void)out_size; (void)ws_size;
    const int*   input  = (const int*)d_in[0];
    const float* latent = (const float*)d_in[1];
    // d_in[2] = code (unused by reference), d_in[3] = lengths (unused),
    // d_in[4] = seq_length (compile-time TT)
    const float* emb    = (const float*)d_in[5];
    const float* Wz_w   = (const float*)d_in[6];
    const float* Wz_b   = (const float*)d_in[7];
    const float* w_ih   = (const float*)d_in[8];
    const float* w_hh   = (const float*)d_in[9];
    const float* b_ih   = (const float*)d_in[10];
    const float* b_hh   = (const float*)d_in[11];
    const float* out_w  = (const float*)d_in[12];
    const float* out_b  = (const float*)d_in[13];
    float* out = (float*)d_out;

    // Workspace carving (256B-aligned)
    char* ws = (char*)d_ws;
    size_t off = 0;
    auto carve = [&](size_t bytes) -> char* {
        char* p = ws + off;
        off = (off + bytes + 255) & ~(size_t)255;
        return p;
    };
    __bf16* embbf  = (__bf16*)carve((size_t)VV * HH * 2);
    __bf16* outwbf = (__bf16*)carve((size_t)VV * HH * 2);
    __bf16* wihbf  = (__bf16*)carve((size_t)3 * HH * HH * 2);
    __bf16* whhbf  = (__bf16*)carve((size_t)3 * HH * HH * 2);
    float*  h      = (float*)carve((size_t)BB * HH * 4);
    __bf16* hbf    = (__bf16*)carve((size_t)BB * HH * 2);
    __bf16* xbf    = (__bf16*)carve((size_t)BB * HH * 2);
    float*  gi     = (float*)carve((size_t)BB * 3 * HH * 4);
    float*  gh     = (float*)carve((size_t)BB * 3 * HH * 4);
    int*    nxt    = (int*)carve((size_t)BB * 4);

    // One-shot bf16 conversion so the whole weight set (~143MB) is L2-resident
    {
        int nVH = VV * HH;                 // 32,768,000
        int nW = 3 * HH * HH;              // 3,145,728
        k_convert_bf16<<<(nVH + 255) / 256, 256, 0, stream>>>(emb, embbf, nVH);
        k_convert_bf16<<<(nVH + 255) / 256, 256, 0, stream>>>(out_w, outwbf, nVH);
        k_convert_bf16<<<(nW + 255) / 256, 256, 0, stream>>>(w_ih, wihbf, nW);
        k_convert_bf16<<<(nW + 255) / 256, 256, 0, stream>>>(w_hh, whhbf, nW);
    }

    const int nBH = BB * HH;  // 32768
    k_init_h0<<<nBH / 256, 256, 0, stream>>>(latent, Wz_w, Wz_b, h, hbf);
    k_init_x0<<<nBH / 256, 256, 0, stream>>>(input, emb, xbf);

    // Sequential decode: argmax feedback forbids parallelism over T
    for (int t = 0; t < TT; ++t) {
        k_gru_gates<<<dim3(12, 2, 2), 128, 0, stream>>>(xbf, hbf, wihbf, whhbf,
                                                        b_ih, b_hh, gi, gh);
        k_gru_pointwise<<<nBH / 256, 256, 0, stream>>>(gi, gh, h, hbf);
        k_logits<<<dim3(125, 2), 128, 0, stream>>>(hbf, outwbf, out_b, out, t);
        k_logsoftmax_argmax<<<BB, 256, 0, stream>>>(out, t, nxt);
        k_gather_x<<<nBH / 256, 256, 0, stream>>>(embbf, nxt, xbf);
    }
}
